// SlotAttention_24601572671955
// MI455X (gfx1250) — compile-verified, matrix operand order checked by reading, not run on hardware
//
#include <hip/hip_runtime.h>
#include <hip/hip_bf16.h>
#include <math.h>

#define BB 64
#define NN 1024
#define DD 64
#define SS 11
#define SP 16
#define HH 128
#define EPS8 1e-8f

typedef __attribute__((ext_vector_type(16))) _Float16 v16h;
typedef __attribute__((ext_vector_type(8)))  float    v8f;
typedef _Float16 h16;

// ---------------- wave / block reduction helpers (wave32) ----------------
__device__ __forceinline__ float wred_sum32(float v) {
#pragma unroll
  for (int off = 16; off > 0; off >>= 1) v += __shfl_xor(v, off, 32);
  return v;
}
__device__ __forceinline__ float wred_max32(float v) {
#pragma unroll
  for (int off = 16; off > 0; off >>= 1) v = fmaxf(v, __shfl_xor(v, off, 32));
  return v;
}
__device__ __forceinline__ float blk256_sum(float* red, int tid, float v) {
  red[tid] = v;
  for (int off = 128; off > 0; off >>= 1) { __syncthreads(); if (tid < off) red[tid] += red[tid + off]; }
  __syncthreads();
  float r = red[0];
  __syncthreads();
  return r;
}
__device__ __forceinline__ float blk256_max(float* red, int tid, float v) {
  red[tid] = v;
  for (int off = 128; off > 0; off >>= 1) { __syncthreads(); if (tid < off) red[tid] = fmaxf(red[tid], red[tid + off]); }
  __syncthreads();
  float r = red[0];
  __syncthreads();
  return r;
}
__device__ __forceinline__ float blk64_sum(float* red, int tid, float v) {
  red[tid] = v;
  for (int off = 32; off > 0; off >>= 1) { __syncthreads(); if (tid < off) red[tid] += red[tid + off]; }
  __syncthreads();
  float r = red[0];
  __syncthreads();
  return r;
}
// 11-wide vector reductions over 256 threads (LDS [256][12])
__device__ __forceinline__ void redsum11(float (*red)[12], int tid) {
  for (int off = 128; off > 0; off >>= 1) {
    __syncthreads();
    if (tid < off) {
#pragma unroll
      for (int s = 0; s < SS; ++s) red[tid][s] += red[tid + off][s];
    }
  }
  __syncthreads();
}
__device__ __forceinline__ void redmax11(float (*red)[12], int tid) {
  for (int off = 128; off > 0; off >>= 1) {
    __syncthreads();
    if (tid < off) {
#pragma unroll
      for (int s = 0; s < SS; ++s) red[tid][s] = fmaxf(red[tid][s], red[tid + off][s]);
    }
  }
  __syncthreads();
}

// ---------------- WMMA fragment helpers (16x16x32 f16 -> f32) ----------------
// A fragment: row-major A[m][k], 16x32 tile, leading dim ld (ISA 7.12.2 layout).
__device__ __forceinline__ v16h frag_a(const h16* A, int ld, int lane) {
  v16h f;
  int m = lane & 15;
  int hi8 = (lane >> 4) << 3;
#pragma unroll
  for (int e = 0; e < 16; ++e) {
    int j = e >> 1;
    int k = ((j >> 2) << 4) + hi8 + ((j & 3) << 1) + (e & 1);
    f[e] = A[(size_t)m * ld + k];
  }
  return f;
}
// B fragment: element B[k][n] fetched as Bp[k*ldk + n*ldn] (mirrored layout).
__device__ __forceinline__ v16h frag_b(const h16* Bp, int ldk, int ldn, int lane) {
  v16h f;
  int n = lane & 15;
  int hi8 = (lane >> 4) << 3;
#pragma unroll
  for (int e = 0; e < 16; ++e) {
    int j = e >> 1;
    int k = ((j >> 2) << 4) + hi8 + ((j & 3) << 1) + (e & 1);
    f[e] = Bp[(size_t)k * ldk + (size_t)n * ldn];
  }
  return f;
}
__device__ __forceinline__ v8f wmma32(v16h a, v16h b, v8f c) {
  return __builtin_amdgcn_wmma_f32_16x16x32_f16(false, a, false, b, (short)0, c, false, false);
}

// ---------------- kernels ----------------

// Wk,Wv -> half
__global__ void k_convw(const float* Wk, const float* Wv, h16* WkH, h16* WvH) {
  int i = blockIdx.x * blockDim.x + threadIdx.x;
  if (i < DD * DD) { WkH[i] = (h16)Wk[i]; WvH[i] = (h16)Wv[i]; }
}

// slots = mu + exp(log_sigma) * slots_init, padded S->SP with zeros
__global__ void k_init_slots(const float* sinit, const float* mu, const float* lsig, float* slots) {
  int idx = blockIdx.x * blockDim.x + threadIdx.x;
  if (idx >= BB * SP * DD) return;
  int d = idx % DD;
  int s = (idx / DD) % SP;
  int b = idx / (SP * DD);
  float v = 0.f;
  if (s < SS) v = mu[d] + expf(lsig[d]) * sinit[((size_t)(b * SS + s)) * DD + d];
  slots[idx] = v;
}

// per-token layernorm + half copy + wi logit. one wave per token.
__global__ void k_prep(const float* x, const float* lng, const float* lnb,
                       const float* wi_w, const float* wi_b,
                       h16* xh, float* logits) {
  int wave = (blockIdx.x * blockDim.x + threadIdx.x) >> 5;
  int lane = threadIdx.x & 31;
  if (wave >= BB * NN) return;
  const float* xp = x + (size_t)wave * DD;
  float v0 = xp[lane], v1 = xp[lane + 32];
  float mean = wred_sum32(v0 + v1) * (1.f / DD);
  float d0 = v0 - mean, d1 = v1 - mean;
  float var = wred_sum32(d0 * d0 + d1 * d1) * (1.f / DD);
  float rstd = rsqrtf(var + 1e-5f);
  float y0 = d0 * rstd * lng[lane] + lnb[lane];
  float y1 = d1 * rstd * lng[lane + 32] + lnb[lane + 32];
  xh[(size_t)wave * DD + lane] = (h16)y0;
  xh[(size_t)wave * DD + lane + 32] = (h16)y1;
  float li = wred_sum32(y0 * wi_w[lane] + y1 * wi_w[lane + 32]);
  if (lane == 0) logits[wave] = li + wi_b[0];
}

// log_a = logit - LSE_N + log(S)
__global__ void k_softmax_a(const float* logits, float* loga) {
  int b = blockIdx.x, tid = threadIdx.x;
  __shared__ float red[256];
  const float* lg = logits + (size_t)b * NN;
  float l[4];
#pragma unroll
  for (int r = 0; r < 4; ++r) l[r] = lg[tid + 256 * r];
  float m = fmaxf(fmaxf(l[0], l[1]), fmaxf(l[2], l[3]));
  float M = blk256_max(red, tid, m);
  float se = 0.f;
#pragma unroll
  for (int r = 0; r < 4; ++r) se += expf(l[r] - M);
  float SE = blk256_sum(red, tid, se);
  float lse = M + logf(SE);
#pragma unroll
  for (int r = 0; r < 4; ++r) loga[(size_t)b * NN + tid + 256 * r] = l[r] - lse + logf((float)SS);
}

// k = x@Wk^T, v = x@Wv^T via WMMA. jobs = mtiles(4096) x ntiles(4) x 2
__global__ void k_gemm_kv(const h16* xh, const h16* WkH, const h16* WvH, h16* kh, h16* vh) {
  int gw = (blockIdx.x * blockDim.x + threadIdx.x) >> 5;
  int lane = threadIdx.x & 31;
  if (gw >= 4096 * 8) return;
  int which = gw & 1;
  int nt = (gw >> 1) & 3;
  int mt = gw >> 3;
  const h16* W = which ? WvH : WkH;
  h16* out = which ? vh : kh;
  const h16* A = xh + (size_t)mt * 16 * DD;
  v8f acc = {};
#pragma unroll
  for (int kk = 0; kk < 2; ++kk) {
    v16h a = frag_a(A + kk * 32, DD, lane);
    v16h bf = frag_b(W + nt * 16 * DD + kk * 32, 1, DD, lane);  // B[k][n]=W[n*DD+k]
    acc = wmma32(a, bf, acc);
  }
  int col = lane & 15, rb = (lane >> 4) << 3;
#pragma unroll
  for (int r = 0; r < 8; ++r)
    out[((size_t)mt * 16 + rb + r) * DD + nt * 16 + col] = (h16)acc[r];
}

__global__ void k_ksq(const h16* kh, float* ksq) {
  int i = blockIdx.x * blockDim.x + threadIdx.x;
  if (i >= BB * NN) return;
  float s = 0.f;
#pragma unroll
  for (int d = 0; d < DD; ++d) { float kv = (float)kh[(size_t)i * DD + d]; s += kv * kv; }
  ksq[i] = s;
}

// per-slot: layernorm, ws logit, q = sn@Wq^T (half), qsq. one 64-thread block per padded slot.
__global__ void k_slotq(const float* slots, const float* lng, const float* lnb,
                        const float* ws_w, const float* ws_b, const float* Wq,
                        h16* qh, float* qsq, float* wslog) {
  int row = blockIdx.x;
  int s = row % SP;
  int t = threadIdx.x;
  if (s >= SS) {
    qh[(size_t)row * DD + t] = (h16)0.f;
    if (t == 0) qsq[row] = 0.f;
    return;
  }
  __shared__ float sn[DD];
  __shared__ float red[64];
  float v = slots[(size_t)row * DD + t];
  float mean = blk64_sum(red, t, v) * (1.f / DD);
  float d = v - mean;
  float var = blk64_sum(red, t, d * d) * (1.f / DD);
  float rstd = rsqrtf(var + 1e-5f);
  float y = d * rstd * lng[t] + lnb[t];
  sn[t] = y;
  __syncthreads();
  float wl = blk64_sum(red, t, y * ws_w[t]);
  if (t == 0) wslog[row] = wl + ws_b[0];
  float q = 0.f;
#pragma unroll 4
  for (int dd = 0; dd < DD; ++dd) q += sn[dd] * Wq[(size_t)t * DD + dd];
  qh[(size_t)row * DD + t] = (h16)q;
  float qs = blk64_sum(red, t, q * q);
  if (t == 0) qsq[row] = qs;
}

// log_b over S=11 per batch; also zero warm-start v. one wave per batch.
__global__ void k_softmax_b(const float* wslog, float* logb, float* v_cur) {
  int b = blockIdx.x, t = threadIdx.x;
  float val = (t < SS) ? wslog[b * SP + t] : -1e30f;
  float M = wred_max32(val);
  float e = (t < SS) ? expf(val - M) : 0.f;
  float SE = wred_sum32(e);
  if (t < SP) {
    v_cur[b * SP + t] = 0.f;
    logb[b * SP + t] = (t < SS) ? (val - M - logf(SE) + logf((float)SS)) : 0.f;
  }
}

__global__ void k_zero_u(float* u_cur) {
  int i = blockIdx.x * blockDim.x + threadIdx.x;
  if (i < BB * NN) u_cur[i] = 0.f;
}

// C[b,n,s] = sqrt(max(ksq+qsq-2*k.q,0)+1e-12) via WMMA; job = (b, mtile)
__global__ void k_cdist(const h16* kh, const h16* qh, const float* ksq, const float* qsq, float* C) {
  int gw = (blockIdx.x * blockDim.x + threadIdx.x) >> 5;
  int lane = threadIdx.x & 31;
  if (gw >= BB * 64) return;
  int mt = gw & 63;
  int b = gw >> 6;
  const h16* A = kh + ((size_t)b * NN + (size_t)mt * 16) * DD;
  const h16* Bq = qh + (size_t)b * SP * DD;
  v8f acc = {};
#pragma unroll
  for (int kk = 0; kk < 2; ++kk) {
    v16h a = frag_a(A + kk * 32, DD, lane);
    v16h bf = frag_b(Bq + kk * 32, 1, DD, lane);  // B[k][n]=q[n*DD+k]
    acc = wmma32(a, bf, acc);
  }
  int col = lane & 15, rb = (lane >> 4) << 3;
#pragma unroll
  for (int r = 0; r < 8; ++r) {
    int n = mt * 16 + rb + r;
    float d2 = ksq[(size_t)b * NN + n] + qsq[b * SP + col] - 2.f * acc[r];
    C[((size_t)b * NN + n) * SP + col] = sqrtf(fmaxf(d2, 0.f) + 1e-12f);
  }
}

// 5 warm-started sinkhorn iterations; stores u_1..u_5, v_0..v_5; updates u_cur/v_cur.
__global__ void k_sink_fwd(const float* C, const float* loga, const float* logb,
                           float* u_cur, float* v_cur, float* u_hist, float* v_hist) {
  int b = blockIdx.x, tid = threadIdx.x;
  __shared__ float red[256][12];
  __shared__ float vsh[SP], lb[SP];
  const float* Cb = C + (size_t)b * NN * SP;
  float p[4][SS], la[4], u[4];
#pragma unroll
  for (int r = 0; r < 4; ++r) {
    int n = tid + 256 * r;
#pragma unroll
    for (int s = 0; s < SS; ++s) p[r][s] = -Cb[(size_t)n * SP + s];
    la[r] = loga[(size_t)b * NN + n];
  }
  if (tid < SP) {
    float vv = v_cur[b * SP + tid];
    vsh[tid] = vv;
    lb[tid] = logb[b * SP + tid];
    v_hist[(b * 6 + 0) * SP + tid] = vv;
  }
  __syncthreads();
  for (int it = 1; it <= 5; ++it) {
#pragma unroll
    for (int r = 0; r < 4; ++r) {
      float mx = -1e30f;
#pragma unroll
      for (int s = 0; s < SS; ++s) mx = fmaxf(mx, p[r][s] + vsh[s]);
      float se = 0.f;
#pragma unroll
      for (int s = 0; s < SS; ++s) se += expf(p[r][s] + vsh[s] - mx);
      u[r] = la[r] - (mx + logf(se));
      u_hist[((size_t)(b * 5 + it - 1)) * NN + tid + 256 * r] = u[r];
    }
    __syncthreads();
#pragma unroll
    for (int s = 0; s < SS; ++s) red[tid][s] = -1e30f;
#pragma unroll
    for (int r = 0; r < 4; ++r)
#pragma unroll
      for (int s = 0; s < SS; ++s) red[tid][s] = fmaxf(red[tid][s], p[r][s] + u[r]);
    redmax11(red, tid);
    float Ms[SS];
#pragma unroll
    for (int s = 0; s < SS; ++s) Ms[s] = red[0][s];
    __syncthreads();
#pragma unroll
    for (int s = 0; s < SS; ++s) {
      float se = 0.f;
#pragma unroll
      for (int r = 0; r < 4; ++r) se += expf(p[r][s] + u[r] - Ms[s]);
      red[tid][s] = se;
    }
    redsum11(red, tid);
    if (tid < SS) vsh[tid] = lb[tid] - (Ms[tid] + logf(red[0][tid]));
    __syncthreads();
    if (tid < SS) v_hist[(b * 6 + it) * SP + tid] = vsh[tid];
  }
#pragma unroll
  for (int r = 0; r < 4; ++r) u_cur[(size_t)b * NN + tid + 256 * r] = u[r];
  if (tid < SS) v_cur[b * SP + tid] = vsh[tid];
}

// reverse-mode grad of H(T) through 5 sinkhorn iterations; C += pbar (MESH_LR=1, gradC=-pbar)
__global__ void k_sink_bwd(float* C, const float* loga, const float* logb,
                           const float* u_hist, const float* v_hist) {
  int b = blockIdx.x, tid = threadIdx.x;
  __shared__ float red[256][12];
  __shared__ float vh[6][SP], lb[SP], gv[SP];
  float* Cb = C + (size_t)b * NN * SP;
  float p[4][SS], pbar[4][SS], la[4], uh[5][4], gu[4];
#pragma unroll
  for (int r = 0; r < 4; ++r) {
    int n = tid + 256 * r;
#pragma unroll
    for (int s = 0; s < SS; ++s) { p[r][s] = -Cb[(size_t)n * SP + s]; pbar[r][s] = 0.f; }
    la[r] = loga[(size_t)b * NN + n];
#pragma unroll
    for (int i = 0; i < 5; ++i) uh[i][r] = u_hist[((size_t)(b * 5 + i)) * NN + n];
  }
  if (tid < SP) {
    lb[tid] = (tid < SS) ? logb[b * SP + tid] : 0.f;
#pragma unroll
    for (int i = 0; i < 6; ++i) vh[i][tid] = (tid < SS) ? v_hist[(b * 6 + i) * SP + tid] : 0.f;
  }
  __syncthreads();
  // seed: T = exp(p+u5+v5), gT = -(log(T+e)-1) - T/(T+e)
#pragma unroll
  for (int s = 0; s < SS; ++s) red[tid][s] = 0.f;
#pragma unroll
  for (int r = 0; r < 4; ++r) {
    gu[r] = 0.f;
#pragma unroll
    for (int s = 0; s < SS; ++s) {
      float T = expf(p[r][s] + uh[4][r] + vh[5][s]);
      float gT = -(logf(T + EPS8) - 1.f) - T / (T + EPS8);
      float w = gT * T;
      pbar[r][s] += w;
      gu[r] += w;
      red[tid][s] += w;
    }
  }
  redsum11(red, tid);
  if (tid < SS) gv[tid] = red[0][tid];
  __syncthreads();
  for (int i = 5; i >= 1; --i) {
    // backprop v_i = log_b - LSE_n(p + u_i): w^v = exp(p + u_i - log_b + v_i)
#pragma unroll
    for (int r = 0; r < 4; ++r)
#pragma unroll
      for (int s = 0; s < SS; ++s) {
        float w = expf(p[r][s] + uh[i - 1][r] - lb[s] + vh[i][s]);
        float t2 = -gv[s] * w;
        pbar[r][s] += t2;
        gu[r] += t2;
      }
    __syncthreads();
    // backprop u_i = log_a - LSE_s(p + v_{i-1}): w^u = exp(p + v_{i-1} - log_a + u_i)
#pragma unroll
    for (int s = 0; s < SS; ++s) red[tid][s] = 0.f;
#pragma unroll
    for (int r = 0; r < 4; ++r)
#pragma unroll
      for (int s = 0; s < SS; ++s) {
        float w = expf(p[r][s] + vh[i - 1][s] - la[r] + uh[i - 1][r]);
        float t2 = -gu[r] * w;
        pbar[r][s] += t2;
        red[tid][s] += t2;
      }
    redsum11(red, tid);
    if (tid < SS) gv[tid] = red[0][tid];  // grad wrt v_{i-1}; dropped at i==1 (stop_gradient)
#pragma unroll
    for (int r = 0; r < 4; ++r) gu[r] = 0.f;
    __syncthreads();
  }
#pragma unroll
  for (int r = 0; r < 4; ++r)
#pragma unroll
    for (int s = 0; s < SS; ++s) Cb[(size_t)(tid + 256 * r) * SP + s] += pbar[r][s];
}

// final sinkhorn(5) on updated C from warm start; emit T (half, transposed, padded) + attn_t (float out)
__global__ void k_sink_attn(const float* C, const float* loga, const float* logb,
                            const float* u_cur, const float* v_cur,
                            h16* Th, float* attn_out) {
  int b = blockIdx.x, tid = threadIdx.x;
  __shared__ float red[256][12];
  __shared__ float vsh[SP], lb[SP];
  const float* Cb = C + (size_t)b * NN * SP;
  float p[4][SS], la[4], u[4];
#pragma unroll
  for (int r = 0; r < 4; ++r) {
    int n = tid + 256 * r;
#pragma unroll
    for (int s = 0; s < SS; ++s) p[r][s] = -Cb[(size_t)n * SP + s];
    la[r] = loga[(size_t)b * NN + n];
  }
  if (tid < SP) { vsh[tid] = v_cur[b * SP + tid]; lb[tid] = logb[b * SP + tid]; }
  __syncthreads();
  for (int it = 1; it <= 5; ++it) {
#pragma unroll
    for (int r = 0; r < 4; ++r) {
      float mx = -1e30f;
#pragma unroll
      for (int s = 0; s < SS; ++s) mx = fmaxf(mx, p[r][s] + vsh[s]);
      float se = 0.f;
#pragma unroll
      for (int s = 0; s < SS; ++s) se += expf(p[r][s] + vsh[s] - mx);
      u[r] = la[r] - (mx + logf(se));
    }
    __syncthreads();
#pragma unroll
    for (int s = 0; s < SS; ++s) red[tid][s] = -1e30f;
#pragma unroll
    for (int r = 0; r < 4; ++r)
#pragma unroll
      for (int s = 0; s < SS; ++s) red[tid][s] = fmaxf(red[tid][s], p[r][s] + u[r]);
    redmax11(red, tid);
    float Ms[SS];
#pragma unroll
    for (int s = 0; s < SS; ++s) Ms[s] = red[0][s];
    __syncthreads();
#pragma unroll
    for (int s = 0; s < SS; ++s) {
      float se = 0.f;
#pragma unroll
      for (int r = 0; r < 4; ++r) se += expf(p[r][s] + u[r] - Ms[s]);
      red[tid][s] = se;
    }
    redsum11(red, tid);
    if (tid < SS) vsh[tid] = lb[tid] - (Ms[tid] + logf(red[0][tid]));
    __syncthreads();
  }
#pragma unroll
  for (int r = 0; r < 4; ++r) {
    int n = tid + 256 * r;
#pragma unroll
    for (int s = 0; s < SS; ++s) {
      float T = expf(p[r][s] + u[r] + vsh[s]);
      Th[((size_t)b * SP + s) * NN + n] = (h16)T;
      attn_out[((size_t)(b * SS + s)) * NN + n] = T;
    }
#pragma unroll
    for (int s = SS; s < SP; ++s) Th[((size_t)b * SP + s) * NN + n] = (h16)0.f;
  }
}

// updates[b, 16, 64] = Th[b,16,1024] @ vh[b,1024,64] via WMMA. job = (b, ntile)
__global__ void k_updates(const h16* Th, const h16* vh, float* upd) {
  int gw = (blockIdx.x * blockDim.x + threadIdx.x) >> 5;
  int lane = threadIdx.x & 31;
  if (gw >= BB * 4) return;
  int b = gw >> 2, nt = gw & 3;
  const h16* A = Th + (size_t)b * SP * NN;
  const h16* Bv = vh + (size_t)b * NN * DD + nt * 16;
  v8f acc = {};
  for (int kt = 0; kt < 32; ++kt) {
    v16h a = frag_a(A + kt * 32, NN, lane);
    v16h bf = frag_b(Bv + (size_t)kt * 32 * DD, DD, 1, lane);  // B[k][n]=v[(kt*32+k)*DD + nt*16+n]
    acc = wmma32(a, bf, acc);
  }
  int col = lane & 15, rb = (lane >> 4) << 3;
#pragma unroll
  for (int r = 0; r < 8; ++r)
    upd[((size_t)b * SP + rb + r) * DD + nt * 16 + col] = acc[r];
}

// GRU + LN + MLP residual per slot (64 threads per slot)
__global__ void k_gru_mlp(const float* upd, float* slots,
                          const float* w_ih, const float* w_hh,
                          const float* b_ih, const float* b_hh,
                          const float* fc1_w, const float* fc1_b,
                          const float* fc2_w, const float* fc2_b,
                          const float* lng, const float* lnb, float* out_slots) {
  int b = blockIdx.x / SS, s = blockIdx.x % SS;
  int t = threadIdx.x;
  int row = b * SP + s;
  __shared__ float xs[DD], hs[DD], pre[DD], hid[HH], red[64];
  xs[t] = upd[(size_t)row * DD + t];
  hs[t] = slots[(size_t)row * DD + t];
  __syncthreads();
  float gi[3], gh[3];
#pragma unroll
  for (int g = 0; g < 3; ++g) {
    float si = b_ih[g * DD + t], sh = b_hh[g * DD + t];
#pragma unroll 4
    for (int d = 0; d < DD; ++d) {
      si += xs[d] * w_ih[(size_t)(g * DD + t) * DD + d];
      sh += hs[d] * w_hh[(size_t)(g * DD + t) * DD + d];
    }
    gi[g] = si; gh[g] = sh;
  }
  float rr = 1.f / (1.f + expf(-(gi[0] + gh[0])));
  float zz = 1.f / (1.f + expf(-(gi[1] + gh[1])));
  float nn = tanhf(gi[2] + rr * gh[2]);
  float hnew = (1.f - zz) * nn + zz * hs[t];
  float mean = blk64_sum(red, t, hnew) * (1.f / DD);
  float d0 = hnew - mean;
  float var = blk64_sum(red, t, d0 * d0) * (1.f / DD);
  float rstd = rsqrtf(var + 1e-5f);
  pre[t] = d0 * rstd * lng[t] + lnb[t];
  __syncthreads();
#pragma unroll
  for (int j = t; j < HH; j += DD) {
    float a = fc1_b[j];
#pragma unroll 4
    for (int d = 0; d < DD; ++d) a += pre[d] * fc1_w[(size_t)j * DD + d];
    hid[j] = fmaxf(a, 0.f);
  }
  __syncthreads();
  float o = fc2_b[t];
#pragma unroll 4
  for (int d = 0; d < HH; ++d) o += hid[d] * fc2_w[(size_t)t * HH + d];
  float res = hnew + o;
  slots[(size_t)row * DD + t] = res;
  out_slots[(size_t)(b * SS + s) * DD + t] = res;
}

// ---------------- host launch ----------------
extern "C" void kernel_launch(void* const* d_in, const int* in_sizes, int n_in,
                              void* d_out, int out_size, void* d_ws, size_t ws_size,
                              hipStream_t stream) {
  const float* x        = (const float*)d_in[0];
  const float* sinit    = (const float*)d_in[1];
  const float* mu       = (const float*)d_in[2];
  const float* lsig     = (const float*)d_in[3];
  const float* Wq       = (const float*)d_in[4];
  const float* Wk       = (const float*)d_in[5];
  const float* Wv       = (const float*)d_in[6];
  const float* w_ih     = (const float*)d_in[7];
  const float* w_hh     = (const float*)d_in[8];
  const float* b_ih     = (const float*)d_in[9];
  const float* b_hh     = (const float*)d_in[10];
  const float* fc1_w    = (const float*)d_in[11];
  const float* fc1_b    = (const float*)d_in[12];
  const float* fc2_w    = (const float*)d_in[13];
  const float* fc2_b    = (const float*)d_in[14];
  const float* ln_in_g  = (const float*)d_in[15];
  const float* ln_in_b  = (const float*)d_in[16];
  const float* ln_sl_g  = (const float*)d_in[17];
  const float* ln_sl_b  = (const float*)d_in[18];
  const float* ln_pr_g  = (const float*)d_in[19];
  const float* ln_pr_b  = (const float*)d_in[20];
  const float* wi_w     = (const float*)d_in[21];
  const float* wi_b     = (const float*)d_in[22];
  const float* ws_w     = (const float*)d_in[23];
  const float* ws_b     = (const float*)d_in[24];
  (void)in_sizes; (void)n_in; (void)out_size; (void)ws_size;

  char* wsp = (char*)d_ws;
  size_t off = 0;
  auto alloc = [&](size_t bytes) -> void* {
    void* p = wsp + off;
    off += (bytes + 255) & ~(size_t)255;
    return p;
  };
  h16* xh    = (h16*)alloc((size_t)BB * NN * DD * 2);
  h16* kh    = (h16*)alloc((size_t)BB * NN * DD * 2);
  h16* vh    = (h16*)alloc((size_t)BB * NN * DD * 2);
  h16* Th    = (h16*)alloc((size_t)BB * SP * NN * 2);
  h16* qh    = (h16*)alloc((size_t)BB * SP * DD * 2);
  h16* WkH   = (h16*)alloc((size_t)DD * DD * 2);
  h16* WvH   = (h16*)alloc((size_t)DD * DD * 2);
  float* logits = (float*)alloc((size_t)BB * NN * 4);
  float* loga   = (float*)alloc((size_t)BB * NN * 4);
  float* ksq    = (float*)alloc((size_t)BB * NN * 4);
  float* u_cur  = (float*)alloc((size_t)BB * NN * 4);
  float* u_hist = (float*)alloc((size_t)5 * BB * NN * 4);
  float* slots  = (float*)alloc((size_t)BB * SP * DD * 4);
  float* qsq    = (float*)alloc((size_t)BB * SP * 4);
  float* wslog  = (float*)alloc((size_t)BB * SP * 4);
  float* logb   = (float*)alloc((size_t)BB * SP * 4);
  float* v_cur  = (float*)alloc((size_t)BB * SP * 4);
  float* v_hist = (float*)alloc((size_t)6 * BB * SP * 4);
  float* C      = (float*)alloc((size_t)BB * NN * SP * 4);
  float* upd    = (float*)alloc((size_t)BB * SP * DD * 4);

  float* out_slots = (float*)d_out;
  float* out_attn  = out_slots + (size_t)BB * SS * DD;

  k_convw<<<16, 256, 0, stream>>>(Wk, Wv, WkH, WvH);
  k_init_slots<<<(BB * SP * DD) / 256, 256, 0, stream>>>(sinit, mu, lsig, slots);
  k_prep<<<(BB * NN) / 8, 256, 0, stream>>>(x, ln_in_g, ln_in_b, wi_w, wi_b, xh, logits);
  k_softmax_a<<<BB, 256, 0, stream>>>(logits, loga);
  k_gemm_kv<<<4096, 256, 0, stream>>>(xh, WkH, WvH, kh, vh);
  k_ksq<<<(BB * NN) / 256, 256, 0, stream>>>(kh, ksq);

  for (int it = 0; it < 3; ++it) {
    k_slotq<<<BB * SP, 64, 0, stream>>>(slots, ln_sl_g, ln_sl_b, ws_w, ws_b, Wq, qh, qsq, wslog);
    k_softmax_b<<<BB, 32, 0, stream>>>(wslog, logb, v_cur);
    k_zero_u<<<(BB * NN) / 256, 256, 0, stream>>>(u_cur);
    k_cdist<<<512, 256, 0, stream>>>(kh, qh, ksq, qsq, C);
    for (int m = 0; m < 4; ++m) {
      k_sink_fwd<<<BB, 256, 0, stream>>>(C, loga, logb, u_cur, v_cur, u_hist, v_hist);
      k_sink_bwd<<<BB, 256, 0, stream>>>(C, loga, logb, u_hist, v_hist);
    }
    k_sink_attn<<<BB, 256, 0, stream>>>(C, loga, logb, u_cur, v_cur, Th, out_attn);
    k_updates<<<32, 256, 0, stream>>>(Th, vh, upd);
    k_gru_mlp<<<BB * SS, 64, 0, stream>>>(upd, slots, w_ih, w_hh, b_ih, b_hh,
                                          fc1_w, fc1_b, fc2_w, fc2_b,
                                          ln_pr_g, ln_pr_b, out_slots);
  }
}
